// PointNetFeaturePropagation_11192684773545
// MI455X (gfx1250) — matrix operand order repaired and checked by reading, not compiled, not run on hardware
//
#include <hip/hip_runtime.h>
#include <cstdint>

// Problem constants (from setup_inputs)
#define B_   8
#define N_   8192
#define S_   2048
#define D1_  128
#define D2_  256
#define CIN_ 384          // D1+D2
#define C1_  256
#define C2_  128
#define M_   (B_*N_)      // 65536 rows

typedef __attribute__((ext_vector_type(16))) __bf16 v16bf;
typedef __attribute__((ext_vector_type(8)))  __bf16 v8bf;
typedef __attribute__((ext_vector_type(8)))  float  v8f;

__device__ __forceinline__ void bsplit(float x, __bf16* hi, __bf16* lo) {
  __bf16 h = (__bf16)x;            // RNE to bf16 (high part)
  *hi = h;
  *lo = (__bf16)(x - (float)h);    // residual (low part)
}

// ---------------------------------------------------------------------------
// Kernel 1: fp32 -> bf16 hi/lo split (weights)
// ---------------------------------------------------------------------------
__global__ void split_kernel(const float* __restrict__ x,
                             __bf16* __restrict__ hi, __bf16* __restrict__ lo,
                             int nelem) {
  int i = blockIdx.x * 256 + threadIdx.x;
  if (i < nelem) bsplit(x[i], &hi[i], &lo[i]);
}

// ---------------------------------------------------------------------------
// Kernel 2: 3-NN search + inverse-distance interpolation + concat.
// Writes feat rows [M_][CIN_] as bf16 hi/lo via 16B v8bf stores.
// Sampled positions staged in 24 KB LDS; uniform LDS reads broadcast.
// ---------------------------------------------------------------------------
__global__ __launch_bounds__(256) void knn_interp_kernel(
    const float* __restrict__ pp,    // [B,3,N]
    const float* __restrict__ spp,   // [B,3,S]
    const float* __restrict__ skipf, // [B,D1,N]
    const float* __restrict__ sfeat, // [B,D2,S]
    __bf16* __restrict__ fHi, __bf16* __restrict__ fLo)
{
  __shared__ float sx[S_], sy[S_], sz[S_];
  const int b   = blockIdx.y;
  const int tid = threadIdx.x;
  const float* sp = spp + (size_t)b * 3 * S_;
  for (int i = tid; i < S_; i += 256) {
    sx[i] = sp[i];
    sy[i] = sp[S_ + i];
    sz[i] = sp[2 * S_ + i];
  }
  __syncthreads();

  const int n = blockIdx.x * 256 + tid;
  const float* p = pp + (size_t)b * 3 * N_;
  const float px = p[n], py = p[N_ + n], pz = p[2 * N_ + n];

  float d0 = 3.4e38f, d1 = 3.4e38f, d2 = 3.4e38f;
  int   i0 = 0, i1 = 0, i2 = 0;
  for (int s = 0; s < S_; ++s) {
    const float dx = px - sx[s], dy = py - sy[s], dz = pz - sz[s];
    const float d = dx * dx + dy * dy + dz * dz;
    if (d < d2) {
      if (d < d1) {
        d2 = d1; i2 = i1;
        if (d < d0) { d1 = d0; i1 = i0; d0 = d; i0 = s; }
        else        { d1 = d;  i1 = s; }
      } else { d2 = d; i2 = s; }
    }
  }
  float w0 = 1.0f / (d0 + 1e-8f);
  float w1 = 1.0f / (d1 + 1e-8f);
  float w2 = 1.0f / (d2 + 1e-8f);
  const float inv = 1.0f / (w0 + w1 + w2);
  w0 *= inv; w1 *= inv; w2 *= inv;

  const size_t rb = ((size_t)b * N_ + n) * CIN_;
  // skip features: coalesced reads across lanes; 16B vector stores per 8 ch
  const float* sk = skipf + (size_t)b * D1_ * N_ + n;
  for (int c0 = 0; c0 < D1_; c0 += 8) {
    v8bf h, l;
#pragma unroll
    for (int j = 0; j < 8; ++j) {
      const float x = sk[(size_t)(c0 + j) * N_];
      const __bf16 hh = (__bf16)x;
      h[j] = hh;
      l[j] = (__bf16)(x - (float)hh);
    }
    *reinterpret_cast<v8bf*>(fHi + rb + c0) = h;
    *reinterpret_cast<v8bf*>(fLo + rb + c0) = l;
  }
  // interpolated features: 3 gathers/channel; sfeat slab (2MB) lives in L2
  const float* sf = sfeat + (size_t)b * D2_ * S_;
  for (int c0 = 0; c0 < D2_; c0 += 8) {
    v8bf h, l;
#pragma unroll
    for (int j = 0; j < 8; ++j) {
      const float* r = sf + (size_t)(c0 + j) * S_;
      const float x = w0 * r[i0] + w1 * r[i1] + w2 * r[i2];
      const __bf16 hh = (__bf16)x;
      h[j] = hh;
      l[j] = (__bf16)(x - (float)hh);
    }
    *reinterpret_cast<v8bf*>(fHi + rb + D1_ + c0) = h;
    *reinterpret_cast<v8bf*>(fLo + rb + D1_ + c0) = l;
  }
}

// ---------------------------------------------------------------------------
// Kernel 3: split-bf16 WMMA GEMM with fused, deterministic BN-stat partials.
//   Y[M][COUT] = A[M][K] * W[COUT][K]^T      (3 WMMAs / 32-K chunk ~ fp32)
// If BN_A: A operand is relu(Yprev*aS[k]+aB[k]) computed on the fly from fp32
// (fuses the inter-layer BN+ReLU+split pass; coefficients cached in LDS).
// Fragment layouts used here, same k order for A and B (wave32):
//   A 16x32 : lane m=(lane&15); elem e: k = (lane>>4)*16 + e
//   B 32x16 : lane n=(lane&15); elem e: k = (lane>>4)*16 + e
//   C/D f32 : lane n=(lane&15); vgpr r: m = (lane>>4)*8 + r
// Block = 8 waves = 32 rows x COUT cols; per-block column sums (val, val^2)
// reduced through LDS in a fixed order -> bitwise deterministic, no atomics.
// ---------------------------------------------------------------------------
template <int K, int COUT, bool BN_A>
__global__ __launch_bounds__(256) void gemm_wmma_kernel(
    const void* __restrict__ A0,       // BN_A ? fp32 Yprev : bf16 hi
    const void* __restrict__ A1,       // BN_A ? unused     : bf16 lo
    const float* __restrict__ aS,      // BN scale per input channel (BN_A)
    const float* __restrict__ aB,      // BN shift per input channel (BN_A)
    const __bf16* __restrict__ WHi, const __bf16* __restrict__ WLo,
    float* __restrict__ Y, float* __restrict__ part)
{
  constexpr int NT  = COUT / 16;   // n-tiles total
  constexpr int TPW = NT / 4;      // n-tiles per wave (4 waves across N)
  __shared__ float s_sum[4][COUT];
  __shared__ float s_sum2[4][COUT];
  __shared__ float s_aS[BN_A ? K : 1];
  __shared__ float s_aB[BN_A ? K : 1];

  const int tid = threadIdx.x;
  if constexpr (BN_A) {
    for (int i = tid; i < K; i += 256) { s_aS[i] = aS[i]; s_aB[i] = aB[i]; }
    __syncthreads();
  }
  const int wave = tid >> 5;
  const int lane = tid & 31;
  const int lrow = lane & 15;
  const int lhi  = lane >> 4;
  const int rowBase = (blockIdx.x * 2 + (wave >> 2)) * 16;
  const int ntBase  = (wave & 3) * TPW;

  v8f acc[TPW] = {};

  for (int kc = 0; kc < K / 32; ++kc) {
    const int kb = kc * 32 + lhi * 16;
    v16bf aHi, aLo;
    if constexpr (BN_A) {
      const float* yr = (const float*)A0 + (size_t)(rowBase + lrow) * K + kb;
#pragma unroll
      for (int j = 0; j < 8; ++j) {
        float v = yr[j] * s_aS[kb + j] + s_aB[kb + j];
        v = v > 0.f ? v : 0.f;
        const __bf16 h = (__bf16)v;
        aHi[j] = h; aLo[j] = (__bf16)(v - (float)h);
        float w = yr[8 + j] * s_aS[kb + 8 + j] + s_aB[kb + 8 + j];
        w = w > 0.f ? w : 0.f;
        const __bf16 h2 = (__bf16)w;
        aHi[8 + j] = h2; aLo[8 + j] = (__bf16)(w - (float)h2);
      }
    } else {
      const size_t aoff = (size_t)(rowBase + lrow) * K + kb;
      const v8bf a0h = *reinterpret_cast<const v8bf*>((const __bf16*)A0 + aoff);
      const v8bf a1h = *reinterpret_cast<const v8bf*>((const __bf16*)A0 + aoff + 8);
      const v8bf a0l = *reinterpret_cast<const v8bf*>((const __bf16*)A1 + aoff);
      const v8bf a1l = *reinterpret_cast<const v8bf*>((const __bf16*)A1 + aoff + 8);
      aHi = __builtin_shufflevector(a0h, a1h, 0,1,2,3,4,5,6,7,8,9,10,11,12,13,14,15);
      aLo = __builtin_shufflevector(a0l, a1l, 0,1,2,3,4,5,6,7,8,9,10,11,12,13,14,15);
    }
#pragma unroll
    for (int t = 0; t < TPW; ++t) {
      const int nout = (ntBase + t) * 16 + lrow;
      const size_t boff = (size_t)nout * K + kc * 32 + lhi * 16;
      const v16bf bHi = *reinterpret_cast<const v16bf*>(WHi + boff);
      const v16bf bLo = *reinterpret_cast<const v16bf*>(WLo + boff);
      acc[t] = __builtin_amdgcn_wmma_f32_16x16x32_bf16(false, aHi, false, bHi,
                                                       (short)0, acc[t], false, false);
      acc[t] = __builtin_amdgcn_wmma_f32_16x16x32_bf16(false, aHi, false, bLo,
                                                       (short)0, acc[t], false, false);
      acc[t] = __builtin_amdgcn_wmma_f32_16x16x32_bf16(false, aLo, false, bHi,
                                                       (short)0, acc[t], false, false);
    }
  }

  // Epilogue: store Y tile + per-block column stats (each LDS slot written once)
  const int rowi = (wave >> 2) * 2 + lhi;
#pragma unroll
  for (int t = 0; t < TPW; ++t) {
    const int ncol = (ntBase + t) * 16 + lrow;
    float s = 0.f, s2 = 0.f;
#pragma unroll
    for (int r = 0; r < 8; ++r) {
      const float v = acc[t][r];
      s += v; s2 += v * v;
      Y[(size_t)(rowBase + lhi * 8 + r) * COUT + ncol] = v;
    }
    s_sum [rowi][ncol] = s;
    s_sum2[rowi][ncol] = s2;
  }
  __syncthreads();
  for (int c = tid; c < COUT; c += 256) {
    part[(size_t)blockIdx.x * 2 * COUT + c] =
        s_sum[0][c] + s_sum[1][c] + s_sum[2][c] + s_sum[3][c];
    part[(size_t)blockIdx.x * 2 * COUT + COUT + c] =
        s_sum2[0][c] + s_sum2[1][c] + s_sum2[2][c] + s_sum2[3][c];
  }
}

// ---------------------------------------------------------------------------
// Kernel 4: finalize BN affine coeffs from P per-block partials.
// out = y*a[c] + sh[c].  Linear bias cancels in train-mode BN -> omitted.
// ---------------------------------------------------------------------------
template <int C>
__global__ void colstats_final_kernel(const float* __restrict__ part, int P,
                                      const float* __restrict__ g,
                                      const float* __restrict__ beta,
                                      float* __restrict__ a,
                                      float* __restrict__ sh) {
  const int c = threadIdx.x;
  double s = 0.0, s2 = 0.0;
  for (int p = 0; p < P; ++p) {
    s  += (double)part[(size_t)p * 2 * C + c];
    s2 += (double)part[(size_t)p * 2 * C + C + c];
  }
  const float mean = (float)(s / (double)M_);
  const float var  = (float)(s2 / (double)M_) - mean * mean;  // biased var
  const float av   = g[c] * rsqrtf(var + 1e-5f);
  a[c]  = av;
  sh[c] = beta[c] - mean * av;
}

// ---------------------------------------------------------------------------
// Kernel 5: BN+ReLU + transpose to [B,C2,N] via LDS tile (32 n x 128 c,
// row stride 129 to avoid bank conflicts). Coalesced on both sides.
// ---------------------------------------------------------------------------
__global__ __launch_bounds__(256) void bnrelu_out_kernel(
    const float* __restrict__ Y, const float* __restrict__ a,
    const float* __restrict__ sh, float* __restrict__ out) {
  __shared__ float tile[32 * 129];
  const int tid = threadIdx.x;
  const int b = blockIdx.y;
  const int n0 = blockIdx.x * 32;
  for (int i = tid; i < 32 * 128; i += 256) {
    const int nl = i >> 7, c = i & 127;
    float v = Y[((size_t)b * N_ + n0 + nl) * C2_ + c] * a[c] + sh[c];
    v = v > 0.f ? v : 0.f;
    tile[nl * 129 + c] = v;
  }
  __syncthreads();
  for (int i = tid; i < 32 * 128; i += 256) {
    const int c = i >> 5, nl = i & 31;
    out[((size_t)b * C2_ + c) * N_ + n0 + nl] = tile[nl * 129 + c];
  }
}

// ---------------------------------------------------------------------------
extern "C" void kernel_launch(void* const* d_in, const int* in_sizes, int n_in,
                              void* d_out, int out_size, void* d_ws, size_t ws_size,
                              hipStream_t stream) {
  (void)in_sizes; (void)n_in; (void)out_size; (void)ws_size;
  const float* pp    = (const float*)d_in[0];
  const float* spp   = (const float*)d_in[1];
  const float* skipf = (const float*)d_in[2];
  const float* sfeat = (const float*)d_in[3];
  const float* W1    = (const float*)d_in[4];
  // d_in[5] = b1 (cancels in train-mode BN)
  const float* g1    = (const float*)d_in[6];
  const float* be1   = (const float*)d_in[7];
  const float* W2    = (const float*)d_in[8];
  // d_in[9] = b2 (cancels)
  const float* g2    = (const float*)d_in[10];
  const float* be2   = (const float*)d_in[11];
  float* out = (float*)d_out;

  char* ws = (char*)d_ws;
  size_t off = 0;
  auto alloc = [&](size_t bytes) -> void* {
    void* p = ws + off;
    off += (bytes + 255) & ~(size_t)255;
    return p;
  };

  constexpr int NBLK = M_ / 32;  // 2048 GEMM blocks -> stat partials
  __bf16* fHi  = (__bf16*)alloc((size_t)M_ * CIN_ * 2);
  __bf16* fLo  = (__bf16*)alloc((size_t)M_ * CIN_ * 2);
  float*  Y1   = (float*)alloc((size_t)M_ * C1_ * 4);
  float*  Y2   = (float*)alloc((size_t)M_ * C2_ * 4);
  __bf16* W1hi = (__bf16*)alloc((size_t)C1_ * CIN_ * 2);
  __bf16* W1lo = (__bf16*)alloc((size_t)C1_ * CIN_ * 2);
  __bf16* W2hi = (__bf16*)alloc((size_t)C2_ * C1_ * 2);
  __bf16* W2lo = (__bf16*)alloc((size_t)C2_ * C1_ * 2);
  float*  part = (float*)alloc((size_t)NBLK * 2 * C1_ * 4);
  float*  a1   = (float*)alloc(C1_ * 4);
  float*  sh1  = (float*)alloc(C1_ * 4);
  float*  a2   = (float*)alloc(C2_ * 4);
  float*  sh2  = (float*)alloc(C2_ * 4);

  // 0) weight hi/lo splits
  split_kernel<<<(C1_ * CIN_ + 255) / 256, 256, 0, stream>>>(W1, W1hi, W1lo, C1_ * CIN_);
  split_kernel<<<(C2_ * C1_ + 255) / 256, 256, 0, stream>>>(W2, W2hi, W2lo, C2_ * C1_);

  // 1) 3-NN + interpolation + concat -> feat (bf16 hi/lo)
  knn_interp_kernel<<<dim3(N_ / 256, B_), 256, 0, stream>>>(pp, spp, skipf, sfeat, fHi, fLo);

  // 2) GEMM1 (+fused BN1 stat partials): [M,384] x [256,384]^T -> Y1 [M,256]
  gemm_wmma_kernel<CIN_, C1_, false><<<NBLK, 256, 0, stream>>>(
      fHi, fLo, nullptr, nullptr, W1hi, W1lo, Y1, part);
  colstats_final_kernel<C1_><<<1, C1_, 0, stream>>>(part, NBLK, g1, be1, a1, sh1);

  // 3) GEMM2 with fused BN1+ReLU+split on A (+fused BN2 stat partials)
  gemm_wmma_kernel<C1_, C2_, true><<<NBLK, 256, 0, stream>>>(
      Y1, nullptr, a1, sh1, W2hi, W2lo, Y2, part);
  colstats_final_kernel<C2_><<<1, C2_, 0, stream>>>(part, NBLK, g2, be2, a2, sh2);

  // 4) BN2+ReLU + transpose to [B,128,N]
  bnrelu_out_kernel<<<dim3(N_ / 32, B_), 256, 0, stream>>>(Y2, a2, sh2, out);
}
